// GATConv_14181982011533
// MI455X (gfx1250) — matrix operand order repaired and checked by reading, not compiled, not run on hardware
//
#include <hip/hip_runtime.h>
#include <hip/hip_bf16.h>

typedef __attribute__((ext_vector_type(16))) _Float16 v16h;
typedef __attribute__((ext_vector_type(8)))  float    v8f;
typedef __attribute__((ext_vector_type(4)))  float    v4f;

#define IN_CH   128
#define HEADS   4
#define OUT_CH  32
#define HO      128      // HEADS*OUT_CH
#define NEG_SLOPE 0.2f
#define GAT_EPS 1e-10f

// Packed W: 32 fragments x 32 lanes x 16 halves = 32 KB
#define WP_HALVES (32 * 32 * 16)

// ---------------------------------------------------------------------------
// Pack W (128x128 f32, row-major K x N) into WMMA B-fragment layout, f16.
// Fragment (kt, nt): K in [kt*32, kt*32+32), N in [nt*16, nt*16+16).
// Lane L: col = nt*16 + (L&15); K start = kt*32 + (L<16 ? 0 : 8);
// halves 0..7 = K start+0..7, halves 8..15 = K start+16..23 (pairs per dword = per VGPR).
// ---------------------------------------------------------------------------
__global__ void pack_w_kernel(const float* __restrict__ W, _Float16* __restrict__ Wp) {
    int t = blockIdx.x * blockDim.x + threadIdx.x;   // 0..1023
    if (t >= 32 * 32) return;
    int frag = t >> 5;            // 0..31  (kt*8 + nt)
    int lane = t & 31;
    int kt = frag >> 3;
    int nt = frag & 7;
    int col    = nt * 16 + (lane & 15);
    int kstart = kt * 32 + ((lane < 16) ? 0 : 8);
    _Float16* dst = Wp + ((size_t)frag * 32 + lane) * 16;
#pragma unroll
    for (int j = 0; j < 16; ++j)
        dst[j] = (_Float16)W[(size_t)(kstart + j + (j >= 8 ? 8 : 0)) * HO + col];
}

// ---------------------------------------------------------------------------
// Zero the accumulators (d_out doubles as `num`, den in workspace).
// ---------------------------------------------------------------------------
__global__ void zero_kernel(float* __restrict__ out, int n_out,
                            float* __restrict__ den, int n_den) {
    int i = blockIdx.x * blockDim.x + threadIdx.x;
    if (i < n_out) out[i] = 0.0f;
    if (i < n_den) den[i] = 0.0f;
}

// ---------------------------------------------------------------------------
// h = x @ W   (N x 128) = (N x 128) x (128 x 128), f16 WMMA, f32 accumulate.
// One wave computes a 16-row x 128-col tile: 8 n-tiles x 4 k-steps of
// v_wmma_f32_16x16x32_f16. Packed W is staged once per workgroup into LDS
// (32 KB of the 320 KB/WGP) so the 8 waves read B fragments via ds_load_b128
// instead of 8x-redundant global traffic.
// ---------------------------------------------------------------------------
__global__ void gemm_kernel(const float* __restrict__ x,
                            const _Float16* __restrict__ Wp,
                            float* __restrict__ h, int n_tiles) {
    __shared__ _Float16 lds_w[WP_HALVES];     // 32 KB

    // Cooperative copy: 256 threads x 8 x 16B. Must run before any wave exits.
    {
        const uint4* s = (const uint4*)Wp;
        uint4*       d = (uint4*)lds_w;
#pragma unroll
        for (int i = 0; i < 8; ++i)
            d[threadIdx.x + 256 * i] = s[threadIdx.x + 256 * i];
    }
    __syncthreads();

    int wave = threadIdx.x >> 5;
    int lane = threadIdx.x & 31;
    int tile = blockIdx.x * (blockDim.x >> 5) + wave;
    if (tile >= n_tiles) return;              // wave-uniform: EXEC stays all-1s

    int row  = tile * 16 + (lane & 15);
    int ksel = (lane < 16) ? 0 : 8;           // A 16-bit layout (ISA 7.12.2)

    // Build the four A fragments (K = 0..127), converted f32 -> f16.
    v16h a[4];
#pragma unroll
    for (int kt = 0; kt < 4; ++kt) {
        const float* p = x + (size_t)row * IN_CH + kt * 32 + ksel;
        v4f p0 = *(const v4f*)(p + 0);
        v4f p1 = *(const v4f*)(p + 4);
        v4f p2 = *(const v4f*)(p + 16);
        v4f p3 = *(const v4f*)(p + 20);
#pragma unroll
        for (int j = 0; j < 4; ++j) {
            a[kt][j]      = (_Float16)p0[j];
            a[kt][4 + j]  = (_Float16)p1[j];
            a[kt][8 + j]  = (_Float16)p2[j];
            a[kt][12 + j] = (_Float16)p3[j];
        }
    }

    int colbase = lane & 15;
    int rowbase = tile * 16 + ((lane < 16) ? 0 : 8);
#pragma unroll
    for (int nt = 0; nt < 8; ++nt) {
        v8f c = {};
#pragma unroll
        for (int kt = 0; kt < 4; ++kt) {
            const v16h b = *((const v16h*)lds_w + ((kt * 8 + nt) * 32 + lane));
            c = __builtin_amdgcn_wmma_f32_16x16x32_f16(
                    false, a[kt], false, b, (short)0, c, false, false);
        }
        int col = nt * 16 + colbase;
#pragma unroll
        for (int r = 0; r < 8; ++r)           // C layout: VGPR r -> row M=r / 8+r
            h[(size_t)(rowbase + r) * HO + col] = c[r];
    }
}

// ---------------------------------------------------------------------------
// Per (node, head): s_src/s_tgt = dot(h[n,hd,:], attention[hd, :32 / 32:]).
// ---------------------------------------------------------------------------
__global__ void score_kernel(const float* __restrict__ h,
                             const float* __restrict__ att,
                             float* __restrict__ s_src,
                             float* __restrict__ s_tgt, int n_nodes) {
    int t = blockIdx.x * blockDim.x + threadIdx.x;
    if (t >= n_nodes * HEADS) return;
    int n  = t >> 2;
    int hd = t & 3;
    const float* hp = h + (size_t)n * HO + hd * OUT_CH;
    const float* as = att + hd * (2 * OUT_CH);
    const float* at = as + OUT_CH;
    float ss = 0.f, st = 0.f;
#pragma unroll 8
    for (int o = 0; o < OUT_CH; ++o) {
        float v = hp[o];
        ss += v * as[o];
        st += v * at[o];
    }
    s_src[t] = ss;
    s_tgt[t] = st;
}

// ---------------------------------------------------------------------------
// One wave per edge. lane handles 4 channels; hd = lane>>3.
// e = exp(-leaky_relu(s_src[src] + s_tgt[tgt])); scatter-add to num/den.
// num/h both ~51MB -> L2-resident; atomics resolve in L2.
// readfirstlane makes the edge id provably uniform so index/score loads
// scalarize instead of issuing 32 identical vector loads.
// ---------------------------------------------------------------------------
__global__ void edge_kernel(const int* __restrict__ edge_index,
                            const float* __restrict__ h,
                            const float* __restrict__ s_src,
                            const float* __restrict__ s_tgt,
                            float* __restrict__ num,     // == d_out
                            float* __restrict__ den, int n_edges) {
    int e = blockIdx.x * (blockDim.x >> 5) + (threadIdx.x >> 5);
    e = __builtin_amdgcn_readfirstlane(e);    // wave-uniform by construction
    if (e >= n_edges) return;
    int lane = threadIdx.x & 31;
    int src = __builtin_amdgcn_readfirstlane(edge_index[e]);
    int tgt = __builtin_amdgcn_readfirstlane(edge_index[n_edges + e]);
    int hd  = lane >> 3;                      // 8 lanes per head

    float score = s_src[src * HEADS + hd] + s_tgt[tgt * HEADS + hd];
    float lr = (score > 0.f) ? score : score * NEG_SLOPE;
    float w  = __expf(-lr);

    v4f hv = *(const v4f*)(h + (size_t)src * HO + lane * 4);
    float* op = num + (size_t)tgt * HO + lane * 4;
    atomicAdd(op + 0, w * hv[0]);
    atomicAdd(op + 1, w * hv[1]);
    atomicAdd(op + 2, w * hv[2]);
    atomicAdd(op + 3, w * hv[3]);
    if ((lane & 7) == 0) atomicAdd(den + tgt * HEADS + hd, w);
}

// ---------------------------------------------------------------------------
// out = num / max(den, EPS) + bias      (in place on d_out)
// ---------------------------------------------------------------------------
__global__ void finalize_kernel(float* __restrict__ out,
                                const float* __restrict__ den,
                                const float* __restrict__ bias, int n_nodes) {
    int i = blockIdx.x * blockDim.x + threadIdx.x;
    if (i >= n_nodes * HO) return;
    int n = i >> 7;
    int c = i & (HO - 1);
    int hd = c >> 5;
    float d = den[n * HEADS + hd];
    d = fmaxf(d, GAT_EPS);
    out[i] = out[i] / d + bias[c];
}

// ---------------------------------------------------------------------------
extern "C" void kernel_launch(void* const* d_in, const int* in_sizes, int n_in,
                              void* d_out, int out_size, void* d_ws, size_t ws_size,
                              hipStream_t stream) {
    const float* x    = (const float*)d_in[0];
    const int*   eidx = (const int*)d_in[1];
    const float* W    = (const float*)d_in[2];
    const float* att  = (const float*)d_in[3];
    const float* bias = (const float*)d_in[4];

    const int N = in_sizes[0] / IN_CH;        // 100000
    const int E = in_sizes[1] / 2;            // 1600000

    // Workspace layout (256B-aligned by construction)
    char* ws = (char*)d_ws;
    float*    h     = (float*)ws;                                   // N*128 f32
    float*    s_src = (float*)(ws + (size_t)N * HO * 4);            // N*4
    float*    s_tgt = s_src + (size_t)N * HEADS;                    // N*4
    float*    den   = s_tgt + (size_t)N * HEADS;                    // N*4
    _Float16* Wp    = (_Float16*)(den + (size_t)N * HEADS);         // 128*128 f16

    float* out = (float*)d_out;

    // 1) Pack W into B-fragment layout
    pack_w_kernel<<<4, 256, 0, stream>>>(W, Wp);

    // 2) Zero accumulators (every call: graph replays reuse buffers)
    {
        int n_out = N * HO, n_den = N * HEADS;
        int blocks = (n_out + 255) / 256;
        zero_kernel<<<blocks, 256, 0, stream>>>(out, n_out, den, n_den);
    }

    // 3) h = x @ W via v_wmma_f32_16x16x32_f16 (one wave per 16-row tile,
    //    B fragments staged through LDS per workgroup)
    {
        int n_tiles = (N + 15) / 16;          // 6250
        int waves_per_block = 8;              // 256 threads
        int blocks = (n_tiles + waves_per_block - 1) / waves_per_block;
        gemm_kernel<<<blocks, 256, 0, stream>>>(x, Wp, h, n_tiles);
    }

    // 4) Attention scores
    {
        int total = N * HEADS;
        score_kernel<<<(total + 255) / 256, 256, 0, stream>>>(h, att, s_src, s_tgt, N);
    }

    // 5) Edge scatter (one wave per edge)
    {
        int waves_per_block = 8;
        int blocks = (E + waves_per_block - 1) / waves_per_block;
        edge_kernel<<<blocks, 256, 0, stream>>>(eidx, h, s_src, s_tgt, out, den, E);
    }

    // 6) Normalize + bias
    {
        int total = N * HO;
        finalize_kernel<<<(total + 255) / 256, 256, 0, stream>>>(out, den, bias, N);
    }
}
